// MultiHeadAttentionQuantum_65481071407855
// MI455X (gfx1250) — matrix operand order repaired and checked by reading, not compiled, not run on hardware
//
#include <hip/hip_runtime.h>
#include <hip/hip_bf16.h>

typedef __bf16 bf16t;
typedef float v4f  __attribute__((ext_vector_type(4)));
typedef float v8f  __attribute__((ext_vector_type(8)));
typedef bf16t v4bf  __attribute__((ext_vector_type(4)));
typedef bf16t v8bf  __attribute__((ext_vector_type(8)));
typedef bf16t v16bf __attribute__((ext_vector_type(16)));

// Problem constants (from reference): B=2048, S=1, E=1024, H=128, DK=8
constexpr int KD = 1024;   // reduction depth for both GEMMs
constexpr int ND = 1024;   // output columns for both GEMMs
constexpr int MD = 2048;   // rows (batch)
constexpr int BM = 128, BN = 128, BK = 32;
constexpr int LDSS = 40;   // padded LDS row stride in bf16 (80B = 5*16B aligned, 20-bank stride)
constexpr int NHEADS = 2048 * 128;

static __device__ inline v16bf cat8(v8bf a, v8bf b) {
    return __builtin_shufflevector(a, b, 0,1,2,3,4,5,6,7,8,9,10,11,12,13,14,15);
}

// bf16x3 split: hi = RN(x), lo = RN(x - float(hi)) (residual is exact in f32)
static __device__ inline void split_bf16(float x, bf16t& h, bf16t& l) {
    h = (bf16t)x;
    l = (bf16t)(x - (float)h);
}

// C = A(MDxKD) * W(NDxKD)^T [+ bias] using bf16x3 split on
// V_WMMA_F32_16X16X32_BF16 (hi*hi + hi*lo + lo*hi, f32 accumulate).
// grid = (ND/BN, MD/BM), block = 256 (8 wave32 waves; 2x4 wave grid; each
// wave computes a 64x32 tile = 4x2 f32 accumulators).
__global__ __launch_bounds__(256)
void gemm_bf16x3_wmma(const float* __restrict__ A,
                      const float* __restrict__ W,
                      const float* __restrict__ bias,
                      float* __restrict__ C)
{
    __shared__ bf16t sAhi[BM * LDSS];
    __shared__ bf16t sAlo[BM * LDSS];
    __shared__ bf16t sBhi[BN * LDSS];
    __shared__ bf16t sBlo[BN * LDSS];

    const int tid   = threadIdx.x;
    const int lane  = tid & 31;
    const int wid   = tid >> 5;       // 0..7
    const int wrow  = wid >> 2;       // 0..1  -> 64-row slab
    const int wcol  = wid & 3;        // 0..3  -> 32-col slab
    const int lhalf = lane >> 4;      // 0/1 : K-half select
    const int lmod  = lane & 15;      // row/col within 16-tile

    const int blockM = blockIdx.y * BM;
    const int blockN = blockIdx.x * BN;

    v8f zero = {};
    v8f acc[4][2];
#pragma unroll
    for (int i = 0; i < 4; ++i)
#pragma unroll
        for (int j = 0; j < 2; ++j) acc[i][j] = zero;

    for (int k0 = 0; k0 < KD; k0 += BK) {
        // Stage 128x32 f32 tiles of A and W, splitting into hi/lo bf16 in LDS.
        // (cvt VALU co-executes with XDL WMMA: 16-bit WMMA is TRANS-tracked.)
#pragma unroll
        for (int t = 0; t < 4; ++t) {
            int idx = tid + t * 256;        // 1024 float4 slots per tile
            int row = idx >> 3;             // 0..127
            int kk  = (idx & 7) << 2;       // 0,4,...,28
            v4f av = *(const v4f*)(A + (size_t)(blockM + row) * KD + k0 + kk);
            v4f wv = *(const v4f*)(W + (size_t)(blockN + row) * KD + k0 + kk);
            v4bf ah, al, wh, wl;
#pragma unroll
            for (int e = 0; e < 4; ++e) {
                bf16t h, l;
                split_bf16(av[e], h, l); ah[e] = h; al[e] = l;
                split_bf16(wv[e], h, l); wh[e] = h; wl[e] = l;
            }
            *(v4bf*)(sAhi + row * LDSS + kk) = ah;
            *(v4bf*)(sAlo + row * LDSS + kk) = al;
            *(v4bf*)(sBhi + row * LDSS + kk) = wh;
            *(v4bf*)(sBlo + row * LDSS + kk) = wl;
        }
        __syncthreads();

        // B fragments (32x16 bf16): lane-half -> contiguous K 0-15 / 16-31.
        v16bf bh[2], bl[2];
#pragma unroll
        for (int ns = 0; ns < 2; ++ns) {
            const int bbase = (wcol * 32 + ns * 16 + lmod) * LDSS + lhalf * 16;
            bh[ns] = cat8(*(const v8bf*)(sBhi + bbase), *(const v8bf*)(sBhi + bbase + 8));
            bl[ns] = cat8(*(const v8bf*)(sBlo + bbase), *(const v8bf*)(sBlo + bbase + 8));
        }

        // A fragments (16x32 bf16): lane-half -> contiguous K 0-15 / 16-31.
#pragma unroll
        for (int ms = 0; ms < 4; ++ms) {
            const int abase = (wrow * 64 + ms * 16 + lmod) * LDSS + lhalf * 16;
            v16bf ahi = cat8(*(const v8bf*)(sAhi + abase), *(const v8bf*)(sAhi + abase + 8));
            v16bf alo = cat8(*(const v8bf*)(sAlo + abase), *(const v8bf*)(sAlo + abase + 8));
#pragma unroll
            for (int ns = 0; ns < 2; ++ns) {
                acc[ms][ns] = __builtin_amdgcn_wmma_f32_16x16x32_bf16(
                    false, ahi, false, bh[ns], (short)0, acc[ms][ns], false, false);
                acc[ms][ns] = __builtin_amdgcn_wmma_f32_16x16x32_bf16(
                    false, ahi, false, bl[ns], (short)0, acc[ms][ns], false, false);
                acc[ms][ns] = __builtin_amdgcn_wmma_f32_16x16x32_bf16(
                    false, alo, false, bh[ns], (short)0, acc[ms][ns], false, false);
            }
        }
        __syncthreads();
    }

    // Epilogue: C/D layout -> lane holds N = lmod, VGPR g holds M = g + 8*lhalf.
#pragma unroll
    for (int ns = 0; ns < 2; ++ns) {
        const int col = blockN + wcol * 32 + ns * 16 + lmod;
        const float bv = bias ? bias[col] : 0.0f;
#pragma unroll
        for (int ms = 0; ms < 4; ++ms) {
            const int rbase = blockM + wrow * 64 + ms * 16 + lhalf * 8;
#pragma unroll
            for (int g = 0; g < 8; ++g)
                C[(size_t)(rbase + g) * ND + col] = acc[ms][ns][g] + bv;
        }
    }
}

// Closed-form 8-qubit circuit: t_j = v_j + rx_j, C_j = cos(t_j);
//   z_0 = C1*...*C7,  z_w = C0*...*Cw (w=1..7).  In-place per 8-float head.
__global__ __launch_bounds__(256)
void quantum_head_kernel(float* __restrict__ v, const float* __restrict__ rx)
{
    int i = blockIdx.x * blockDim.x + threadIdx.x;
    if (i >= NHEADS) return;
    float* p = v + (size_t)i * 8;

    v4f lo = *(const v4f*)(p);
    v4f hi = *(const v4f*)(p + 4);
    float c[8];
#pragma unroll
    for (int j = 0; j < 4; ++j) c[j]     = __cosf(lo[j] + rx[j]);
#pragma unroll
    for (int j = 0; j < 4; ++j) c[4 + j] = __cosf(hi[j] + rx[4 + j]);

    float z[8];
    float r = c[0];
#pragma unroll
    for (int w = 1; w < 8; ++w) { r *= c[w]; z[w] = r; }
    float s = c[1];
#pragma unroll
    for (int w = 2; w < 8; ++w) s *= c[w];
    z[0] = s;

    v4f olo, ohi;
#pragma unroll
    for (int j = 0; j < 4; ++j) { olo[j] = z[j]; ohi[j] = z[4 + j]; }
    *(v4f*)(p)     = olo;
    *(v4f*)(p + 4) = ohi;
}

extern "C" void kernel_launch(void* const* d_in, const int* in_sizes, int n_in,
                              void* d_out, int out_size, void* d_ws, size_t ws_size,
                              hipStream_t stream)
{
    (void)in_sizes; (void)n_in; (void)out_size; (void)ws_size;
    const float* x  = (const float*)d_in[0];
    // d_in[1]=wq, d_in[2]=wk are mathematically dead: softmax over S==1 is 1.0
    const float* wv = (const float*)d_in[3];
    const float* wc = (const float*)d_in[4];
    const float* bc = (const float*)d_in[5];
    const float* rx = (const float*)d_in[6];
    float* V   = (float*)d_ws;    // 2048*1024 f32 = 8 MB scratch, overwritten each call
    float* out = (float*)d_out;

    dim3 grid(ND / BN, MD / BM);  // (8, 16)
    gemm_bf16x3_wmma<<<grid, 256, 0, stream>>>(x, wv, nullptr, V);
    quantum_head_kernel<<<NHEADS / 256, 256, 0, stream>>>(V, rx);
    gemm_bf16x3_wmma<<<grid, 256, 0, stream>>>(V, wc, bc, out);
}
